// BiMambaLM_66236985639169
// MI455X (gfx1250) — compile-verified, matrix operand order checked by reading, not run on hardware
//
#include <hip/hip_runtime.h>
#include <hip/hip_bf16.h>

// ---------------------------------------------------------------------------
// BiMamba LM on MI455X (gfx1250, wave32).
// GEMMs: f16 WMMA (v_wmma_f32_16x16x32_f16), f32 accumulate.
//   Tile staging via Tensor Data Mover (tensor_load_to_lds, TENSORcnt),
//   double-buffered LDS, software pipelined. Fallback: manual staging.
// Scan: fused per-(b,e) recurrence with h[16] in VGPRs, B/C via LDS broadcast.
// ---------------------------------------------------------------------------

#define DD     512
#define NN     16
#define ED     1024
#define DCONV  4
#define DTR    32
#define DEPTH  6
#define VOCAB  32000
#define BB     2
#define LL     512
#define BL     (BB * LL)     // 1024 rows in every GEMM
#define EPSF   1e-5f

typedef __attribute__((ext_vector_type(16))) _Float16     v16h;
typedef __attribute__((ext_vector_type(8)))  float        v8f;
typedef __attribute__((ext_vector_type(4)))  unsigned int u32x4;
typedef __attribute__((ext_vector_type(8)))  int          i32x8;
typedef __attribute__((ext_vector_type(4)))  int          i32x4;

#define MODE_STORE    0
#define MODE_SOFTPLUS 1
#define MODE_ACCUM    2

#ifndef USE_TDM
#if defined(__has_builtin)
#if __has_builtin(__builtin_amdgcn_tensor_load_to_lds) && __has_builtin(__builtin_amdgcn_s_wait_tensorcnt)
#define USE_TDM 1
#else
#define USE_TDM 0
#endif
#else
#define USE_TDM 0
#endif
#endif

// ---------------------------------------------------------------------------
// Embedding gather: x[row,:] = E[tokens[row],:]
// ---------------------------------------------------------------------------
__global__ __launch_bounds__(128) void embed_kernel(const int* __restrict__ tokens,
                                                    const float* __restrict__ E,
                                                    float* __restrict__ x) {
    int row = blockIdx.x;                 // 0 .. BL-1
    int tok = tokens[row];
    const float* src = E + (size_t)tok * DD;
    float* dst = x + (size_t)row * DD;
    for (int i = threadIdx.x; i < DD; i += 128) dst[i] = src[i];
}

// ---------------------------------------------------------------------------
// RMSNorm per row of length D=512
// ---------------------------------------------------------------------------
__global__ __launch_bounds__(128) void rmsnorm_kernel(const float* __restrict__ x,
                                                      const float* __restrict__ w,
                                                      float* __restrict__ out) {
    __shared__ float s[128];
    int row = blockIdx.x;
    const float* xr = x + (size_t)row * DD;
    float acc = 0.f;
    for (int i = threadIdx.x; i < DD; i += 128) { float v = xr[i]; acc += v * v; }
    s[threadIdx.x] = acc;
    __syncthreads();
    for (int off = 64; off > 0; off >>= 1) {
        if (threadIdx.x < off) s[threadIdx.x] += s[threadIdx.x + off];
        __syncthreads();
    }
    float scale = rsqrtf(s[0] / (float)DD + EPSF);
    for (int i = threadIdx.x; i < DD; i += 128)
        out[(size_t)row * DD + i] = xr[i] * scale * w[i];
}

// ---------------------------------------------------------------------------
// TDM issue: DMA a 2D tile (rows x 32 f32, row stride = strideElems) from
// global memory into LDS at byte offset ldsoff. D# layout per CDNA5 ISA ch.8.
// ---------------------------------------------------------------------------
#if USE_TDM
__device__ __forceinline__ void tdm_issue_tile(const float* gsrc, unsigned ldsoff,
                                               unsigned rows, unsigned strideElems) {
    unsigned long long ga = (unsigned long long)(size_t)gsrc;
    unsigned td0 = strideElems;          // tensor dim0 (>= tile width 32)
    unsigned td1 = 1u << 20;             // tensor dim1: large, no clipping
    u32x4 g0;
    g0[0] = 1u;                                              // count=1, user D#
    g0[1] = ldsoff;                                          // lds_addr (bytes)
    g0[2] = (unsigned)ga;                                    // global_addr[31:0]
    g0[3] = (unsigned)((ga >> 32) & 0x1FFFFFFu) | (2u << 30); // addr[56:32]|type=2
    i32x8 g1;
    g1[0] = (int)(2u << 16);                                 // data_size = 4B
    g1[1] = (int)(td0 << 16);                                // tensor_dim0 lo16
    g1[2] = (int)((td0 >> 16) | ((td1 & 0xFFFFu) << 16));    // dim0 hi / dim1 lo
    g1[3] = (int)((td1 >> 16) | (32u << 16));                // dim1 hi / tile_dim0=32
    g1[4] = (int)rows;                                       // tile_dim1, tile_dim2=0
    g1[5] = (int)strideElems;                                // tensor_dim0_stride lo32
    g1[6] = 0;                                               // stride hi16, dim1_stride lo
    g1[7] = 0;
    i32x4 z4 = {0, 0, 0, 0};
    i32x8 z8 = {0, 0, 0, 0, 0, 0, 0, 0};
    __builtin_amdgcn_tensor_load_to_lds(g0, g1, z4, z4, z8, 0);
}
#endif

// f32 LDS row -> f16 WMMA fragment (CDNA5 16-bit layout: lane half h reads
// K = {8h..8h+7} and {16+8h..16+8h+7}); p points at row + 8h.
__device__ __forceinline__ v16h frag_from_f32(const float* __restrict__ p) {
    v16h f;
    #pragma unroll
    for (int i = 0; i < 8; i++) {
        f[i]     = (_Float16)p[i];
        f[8 + i] = (_Float16)p[16 + i];
    }
    return f;
}

// ---------------------------------------------------------------------------
// WMMA GEMM:  C[M,N] (+)= act( A[M,K] * W[N,K]^T (+ bias) )
//   block = 256 threads (8 waves), tile = 128(M) x 64(N), K-step = 32.
//   wave (wy,wx) owns a 32x32 patch = 2x2 fragments of 16x16.
// Requirements: M%128==0, N%64==0, K%32==0, W row stride == K.
// ---------------------------------------------------------------------------
template <int MODE>
__global__ __launch_bounds__(256) void gemm_wmma_kernel(
    const float* __restrict__ A, const float* __restrict__ W,
    float* __restrict__ C, const float* __restrict__ bias,
    int K, int lda, int ldc) {

    __shared__ float sA[2][128 * 32];
    __shared__ float sB[2][64 * 32];

    const int tid  = threadIdx.x;
    const int lane = tid & 31;
    const int wv   = tid >> 5;      // 0..7
    const int wy   = wv & 3;        // row group (4 x 32 rows)
    const int wx   = wv >> 2;       // col group (2 x 32 cols)
    const int lm   = lane & 15;
    const int h    = lane >> 4;

    const int rowBase = blockIdx.y * 128;
    const int colBase = blockIdx.x * 64;
    const int nk = K >> 5;

    v8f acc00 = {}; v8f acc01 = {}; v8f acc10 = {}; v8f acc11 = {};

#if USE_TDM
    const unsigned ldsA0 = (unsigned)(size_t)(const void*)&sA[0][0];
    const unsigned ldsA1 = (unsigned)(size_t)(const void*)&sA[1][0];
    const unsigned ldsB0 = (unsigned)(size_t)(const void*)&sB[0][0];
    const unsigned ldsB1 = (unsigned)(size_t)(const void*)&sB[1][0];
    if (wv == 0) {   // prologue: DMA tiles for k-step 0 into buffer 0
        tdm_issue_tile(A + (size_t)rowBase * lda, ldsA0, 128, (unsigned)lda);
        tdm_issue_tile(W + (size_t)colBase * K,   ldsB0, 64,  (unsigned)K);
    }
    for (int i = 0; i < nk; i++) {
        const int buf = i & 1;
        if (wv == 0) {
            if (i + 1 < nk) {   // pipeline: DMA next tiles into other buffer
                tdm_issue_tile(A + (size_t)rowBase * lda + (i + 1) * 32,
                               buf ? ldsA0 : ldsA1, 128, (unsigned)lda);
                tdm_issue_tile(W + (size_t)colBase * K + (i + 1) * 32,
                               buf ? ldsB0 : ldsB1, 64, (unsigned)K);
                __builtin_amdgcn_s_wait_tensorcnt(2);  // in-order: pair i done
            } else {
                __builtin_amdgcn_s_wait_tensorcnt(0);
            }
        }
        __syncthreads();
        {
            const float* pa0 = &sA[buf][(wy * 32 +      lm) * 32 + h * 8];
            const float* pa1 = &sA[buf][(wy * 32 + 16 + lm) * 32 + h * 8];
            const float* pb0 = &sB[buf][(wx * 32 +      lm) * 32 + h * 8];
            const float* pb1 = &sB[buf][(wx * 32 + 16 + lm) * 32 + h * 8];
            v16h a0 = frag_from_f32(pa0), a1 = frag_from_f32(pa1);
            v16h b0 = frag_from_f32(pb0), b1 = frag_from_f32(pb1);
            acc00 = __builtin_amdgcn_wmma_f32_16x16x32_f16(false, a0, false, b0, (short)0, acc00, false, false);
            acc01 = __builtin_amdgcn_wmma_f32_16x16x32_f16(false, a0, false, b1, (short)0, acc01, false, false);
            acc10 = __builtin_amdgcn_wmma_f32_16x16x32_f16(false, a1, false, b0, (short)0, acc10, false, false);
            acc11 = __builtin_amdgcn_wmma_f32_16x16x32_f16(false, a1, false, b1, (short)0, acc11, false, false);
        }
        __syncthreads();
    }
#else
    // Fallback: cooperative manual staging, single buffered.
    const int ar = tid >> 1, ah = (tid & 1) * 16;   // A: 2 thr/row, 16 floats
    const int br = tid >> 2, bq = (tid & 3) * 8;    // B: 4 thr/row, 8 floats
    for (int i = 0; i < nk; i++) {
        const int k0 = i * 32;
        {
            const float4* src = (const float4*)(A + (size_t)(rowBase + ar) * lda + k0 + ah);
            #pragma unroll
            for (int v = 0; v < 4; v++) *(float4*)&sA[0][ar * 32 + ah + v * 4] = src[v];
        }
        {
            const float4* src = (const float4*)(W + (size_t)(colBase + br) * K + k0 + bq);
            #pragma unroll
            for (int v = 0; v < 2; v++) *(float4*)&sB[0][br * 32 + bq + v * 4] = src[v];
        }
        __syncthreads();
        const float* pa0 = &sA[0][(wy * 32 +      lm) * 32 + h * 8];
        const float* pa1 = &sA[0][(wy * 32 + 16 + lm) * 32 + h * 8];
        const float* pb0 = &sB[0][(wx * 32 +      lm) * 32 + h * 8];
        const float* pb1 = &sB[0][(wx * 32 + 16 + lm) * 32 + h * 8];
        v16h a0 = frag_from_f32(pa0), a1 = frag_from_f32(pa1);
        v16h b0 = frag_from_f32(pb0), b1 = frag_from_f32(pb1);
        acc00 = __builtin_amdgcn_wmma_f32_16x16x32_f16(false, a0, false, b0, (short)0, acc00, false, false);
        acc01 = __builtin_amdgcn_wmma_f32_16x16x32_f16(false, a0, false, b1, (short)0, acc01, false, false);
        acc10 = __builtin_amdgcn_wmma_f32_16x16x32_f16(false, a1, false, b0, (short)0, acc10, false, false);
        acc11 = __builtin_amdgcn_wmma_f32_16x16x32_f16(false, a1, false, b1, (short)0, acc11, false, false);
        __syncthreads();
    }
#endif

    // ---- epilogue: C/D layout: lane lm = col, VGPR j -> row 8*h + j ----
    const int col0 = colBase + wx * 32 + lm;
    const int row0 = rowBase + wy * 32 + 8 * h;

    #pragma unroll
    for (int fm = 0; fm < 2; fm++) {
        #pragma unroll
        for (int fn = 0; fn < 2; fn++) {
            v8f acc = (fm == 0) ? (fn == 0 ? acc00 : acc01)
                                : (fn == 0 ? acc10 : acc11);
            const int col = col0 + fn * 16;
            const int row = row0 + fm * 16;
            #pragma unroll
            for (int j = 0; j < 8; j++) {
                float v = acc[j];
                size_t idx = (size_t)(row + j) * ldc + col;
                if constexpr (MODE == MODE_SOFTPLUS) {
                    v += bias[col];
                    v = (v > 20.f) ? v : __logf(1.f + __expf(v));
                    C[idx] = v;
                } else if constexpr (MODE == MODE_ACCUM) {
                    C[idx] += v;
                } else {
                    C[idx] = v;
                }
            }
        }
    }
}

// ---------------------------------------------------------------------------
// Depthwise causal conv (DCONV=4) + bias + SiLU.
//   dir=0: y[t] = sum_j w[j]*x[t-3+j]   (causal)
//   dir=1: y[t] = sum_j w[j]*x[t+3-j]   (anti-causal == conv on flipped seq)
// ---------------------------------------------------------------------------
__global__ __launch_bounds__(256) void conv_silu_kernel(
    const float* __restrict__ xz, const float* __restrict__ cw,
    const float* __restrict__ cb, float* __restrict__ out, int dir) {
    int idx = blockIdx.x * 256 + threadIdx.x;   // over B*L*ED
    int e = idx & (ED - 1);
    int t = (idx >> 10) & (LL - 1);
    int b = idx >> 19;
    float acc = cb[e];
    #pragma unroll
    for (int j = 0; j < DCONV; j++) {
        int tt = (dir == 0) ? (t - (DCONV - 1) + j) : (t + (DCONV - 1) - j);
        if (tt >= 0 && tt < LL)
            acc += cw[e * DCONV + j] * xz[((size_t)(b * LL + tt)) * (2 * ED) + e];
    }
    out[idx] = acc / (1.f + __expf(-acc));      // silu
}

// ---------------------------------------------------------------------------
// Selective scan, fused: h[n] = exp(delta*A[n])*h[n] + delta*B[n]*xs
//   y = (sum_n h[n]*C[n] + Dp*xs) * silu(z)
// One thread per (b,e); B/C (32 floats per (b,t)) broadcast via LDS.
// dir=1 runs t = L-1 .. 0 (backward pass on unflipped storage).
// ---------------------------------------------------------------------------
__global__ __launch_bounds__(256) void scan_kernel(
    const float* __restrict__ delta, const float* __restrict__ xs,
    const float* __restrict__ dbl, const float* __restrict__ xz,
    const float* __restrict__ Alog, const float* __restrict__ Dp,
    float* __restrict__ y, int dir) {

    __shared__ float sBC[2 * NN];               // B then C
    int b = blockIdx.x >> 2;                    // ED/256 = 4 chunks
    int e = (blockIdx.x & 3) * 256 + threadIdx.x;

    float A[NN], hst[NN];
    #pragma unroll
    for (int n = 0; n < NN; n++) {
        A[n] = -__expf(Alog[(size_t)e * NN + n]);
        hst[n] = 0.f;
    }
    float dpe = Dp[e];

    for (int s = 0; s < LL; s++) {
        int t = dir ? (LL - 1 - s) : s;
        size_t base = (size_t)(b * LL + t);
        if (threadIdx.x < 2 * NN)
            sBC[threadIdx.x] = dbl[base * (DTR + 2 * NN) + DTR + threadIdx.x];
        __syncthreads();

        float d  = delta[base * ED + e];
        float xv = xs[base * ED + e];
        float zv = xz[base * (2 * ED) + ED + e];
        float acc = 0.f;
        #pragma unroll
        for (int n = 0; n < NN; n++) {
            float dA = __expf(d * A[n]);
            hst[n] = dA * hst[n] + (d * sBC[n]) * xv;
            acc += hst[n] * sBC[NN + n];
        }
        float sig = 1.f / (1.f + __expf(-zv));
        y[base * ED + e] = (acc + dpe * xv) * (zv * sig);
        __syncthreads();
    }
}

// ---------------------------------------------------------------------------
// Host orchestration
// ---------------------------------------------------------------------------
extern "C" void kernel_launch(void* const* d_in, const int* in_sizes, int n_in,
                              void* d_out, int out_size, void* d_ws, size_t ws_size,
                              hipStream_t stream) {
    const int*   tokens     = (const int*)  d_in[0];
    const float* E          = (const float*)d_in[1];
    const float* norm_w     = (const float*)d_in[2];
    const float* W_in       = (const float*)d_in[3];
    const float* conv_w     = (const float*)d_in[4];
    const float* conv_b     = (const float*)d_in[5];
    const float* W_xp       = (const float*)d_in[6];
    const float* W_dt       = (const float*)d_in[7];
    const float* b_dt       = (const float*)d_in[8];
    const float* A_log      = (const float*)d_in[9];
    const float* Dparam     = (const float*)d_in[10];
    const float* W_out      = (const float*)d_in[11];
    const float* out_norm_w = (const float*)d_in[12];
    float* logits = (float*)d_out;

    // workspace carve-up (floats); ~24.3 MB total
    float* p     = (float*)d_ws;
    float* x     = p; p += (size_t)BL * DD;          // residual stream
    float* xn    = p; p += (size_t)BL * DD;          // rmsnorm output
    float* xz    = p; p += (size_t)BL * 2 * ED;      // in-proj (xs | z)
    float* xsc   = p; p += (size_t)BL * ED;          // conv+silu output
    float* dbl   = p; p += (size_t)BL * (DTR + 2 * NN);
    float* delta = p; p += (size_t)BL * ED;
    float* yb    = p; p += (size_t)BL * ED;

    embed_kernel<<<BL, 128, 0, stream>>>(tokens, E, x);

    for (int l = 0; l < DEPTH; l++) {
        rmsnorm_kernel<<<BL, 128, 0, stream>>>(x, norm_w + (size_t)l * DD, xn);
        for (int d = 0; d < 2; d++) {
            const size_t o = (size_t)(l * 2 + d);
            // xz = xn @ W_in^T      (1024 x 2048 x 512)
            gemm_wmma_kernel<MODE_STORE><<<dim3(2 * ED / 64, BL / 128), 256, 0, stream>>>(
                xn, W_in + o * (2 * ED) * DD, xz, nullptr, DD, DD, 2 * ED);
            // depthwise conv + silu
            conv_silu_kernel<<<(BL * ED) / 256, 256, 0, stream>>>(
                xz, conv_w + o * ED * DCONV, conv_b + o * ED, xsc, d);
            // dbl = xs @ W_xp^T     (1024 x 64 x 1024)
            gemm_wmma_kernel<MODE_STORE><<<dim3((DTR + 2 * NN) / 64, BL / 128), 256, 0, stream>>>(
                xsc, W_xp + o * (DTR + 2 * NN) * ED, dbl, nullptr, ED, ED, DTR + 2 * NN);
            // delta = softplus(dtr @ W_dt^T + b_dt)   (1024 x 1024 x 32)
            gemm_wmma_kernel<MODE_SOFTPLUS><<<dim3(ED / 64, BL / 128), 256, 0, stream>>>(
                dbl, W_dt + o * ED * DTR, delta, b_dt + o * ED, DTR, DTR + 2 * NN, ED);
            // fused selective scan -> yb
            scan_kernel<<<BB * (ED / 256), 256, 0, stream>>>(
                delta, xsc, dbl, xz,
                A_log + o * ED * NN, Dparam + o * ED, yb, d);
            // x += yb @ W_out^T     (1024 x 512 x 1024)
            gemm_wmma_kernel<MODE_ACCUM><<<dim3(DD / 64, BL / 128), 256, 0, stream>>>(
                yb, W_out + o * DD * ED, x, nullptr, ED, ED, DD);
        }
    }

    // final norm + vocab projection: logits = xn @ E^T (1024 x 32000 x 512)
    rmsnorm_kernel<<<BL, 128, 0, stream>>>(x, out_norm_w, xn);
    gemm_wmma_kernel<MODE_STORE><<<dim3(VOCAB / 64, BL / 128), 256, 0, stream>>>(
        xn, E, logits, nullptr, DD, DD, VOCAB);
}
